// CorrelationLayer_48773648613915
// MI455X (gfx1250) — matrix operand order repaired and checked by reading, not compiled, not run on hardware
//
#include <hip/hip_runtime.h>

typedef __attribute__((ext_vector_type(16))) _Float16 v16h;
typedef __attribute__((ext_vector_type(2)))  _Float16 h2;
typedef __attribute__((ext_vector_type(8)))  float    v8f;
typedef __attribute__((ext_vector_type(4)))  float    v4f;

#define MAXD  4
#define ND    9               // displacements per axis
#define NC    128             // channels (K)
#define HH    96
#define WW    192
#define NB    8
#define WT    16              // pixel tile (M)
#define BW    32              // staged window cols (24 used + zero pad)
#define HB    8               // output rows per workgroup == waves
#define NROWS (HB + 2*MAXD)   // 16 f2 rows per workgroup
#define GS    36              // padded G row stride (floats) -> conflict-free
#define BBLK  536             // halfs per K-block: 32*16 + 3*8 pad (16B-aligned)
#define HWSZ  (HH * WW)

// B slot: block (c>>4) * BBLK + n*16 + 8*(n>>3) + (c&15)
__device__ __forceinline__ int bslot(int c, int n) {
    return (c >> 4) * BBLK + n * 16 + 8 * (n >> 3) + (c & 15);
}

__global__ __launch_bounds__(256)
void CorrelationLayer_48773648613915_kernel(const float* __restrict__ f1,
                                            const float* __restrict__ f2,
                                            float* __restrict__ out)
{
    __shared__ _Float16 Alds[HB][WT * NC];            // 32 KB transpose buffer
    __shared__ alignas(32) _Float16 Bf[8 * BBLK];     // 8.4 KB fragment-ready B
    __shared__ float Glds[HB][WT * GS];               // 18 KB extraction scratch

    const int tid  = threadIdx.x;
    const int lane = tid & 31;
    const int wv   = tid >> 5;
    const int w0   = blockIdx.x * WT;
    const int h0   = blockIdx.y * HB;
    const int b    = blockIdx.z;
    const size_t chs = (size_t)HWSZ;

    // ---- stage 8 A tiles (f32 -> f16), float4 over m ----
    {
        const float* f1b = f1 + (size_t)b * NC * chs;
        for (int idx = tid; idx < HB * NC * (WT / 4); idx += 256) {
            const int hA  = idx >> 9;
            const int rem = idx & 511;
            const int c   = rem >> 2;
            const int mq  = (rem & 3) << 2;
            const v4f v = *(const v4f*)(f1b + (size_t)c * chs +
                                        (size_t)(h0 + hA) * WW + w0 + mq);
            _Float16* dst = &Alds[hA][0];
            dst[(mq + 0) * NC + c] = (_Float16)v[0];
            dst[(mq + 1) * NC + c] = (_Float16)v[1];
            dst[(mq + 2) * NC + c] = (_Float16)v[2];
            dst[(mq + 3) * NC + c] = (_Float16)v[3];
        }
    }

    const int row16 = lane & 15;
    const int hi    = lane >> 4;

    __syncthreads();

    // ---- hoist this wave's A fragments into registers (constant all rows) ----
    v16h afrag[4];
    {
        const _Float16* Arow = &Alds[wv][row16 * NC];
        #pragma unroll
        for (int kc = 0; kc < 4; ++kc) {
            #pragma unroll
            for (int j = 0; j < 8; ++j) {
                const int kb = 2 * j + 16 * hi;
                afrag[kc][2 * j]     = Arow[kc * 32 + kb];
                afrag[kc][2 * j + 1] = Arow[kc * 32 + kb + 1];
            }
        }
    }

    // ---- precompute staging descriptors (2 units/thread, fixed per thread) ----
    // unit: channel pair cp (c=2cp,2cp+1), quad nq; global 128B-coalesced loads,
    // packed b32 LDS stores hit 32 distinct banks.
    long long goff[2];  int dslot[2];  int nqv[2];  int w2v[2];  bool qfull[2], qany[2];
    #pragma unroll
    for (int i = 0; i < 2; ++i) {
        const int u  = tid + 256 * i;
        const int cp = u >> 3;
        const int nq = (u & 7) << 2;
        const int w2 = w0 - MAXD + nq;
        nqv[i]   = nq;
        w2v[i]   = w2;
        goff[i]  = (long long)(2 * cp) * (long long)chs + w2;
        dslot[i] = bslot(2 * cp, nq);
        qfull[i] = (nq < (WT + 2 * MAXD)) && (w2 >= 0) && (w2 + 3 < WW);
        qany[i]  = (nq < (WT + 2 * MAXD)) && (w2 + 3 >= 0) && (w2 < WW);
    }

    // ---- precompute extraction offsets (per lane, fixed) ----
    int gofs[5], oofs[5];
    #pragma unroll
    for (int i = 0; i < 5; ++i) {
        const int e  = lane + 32 * i;
        const int dx = e >> 4;
        const int m  = e & 15;
        gofs[i] = m * GS + m + dx;
        oofs[i] = dx * HWSZ + m;
    }
    float* const outbase = out + (((size_t)b * (ND * ND)) * HH + (h0 + wv)) * WW + w0;
    const float invc = 1.0f / (float)NC;
    const float* const f2b = f2 + (size_t)b * NC * chs;

    for (int k = 0; k < NROWS; ++k) {
        const int  r     = h0 - MAXD + k;
        const bool rowok = (r >= 0) && (r < HH);

        // ---- stage f2 row: conflict-free packed b32 stores ----
        #pragma unroll
        for (int i = 0; i < 2; ++i) {
            v4f va = {0.0f, 0.0f, 0.0f, 0.0f};
            v4f vb = {0.0f, 0.0f, 0.0f, 0.0f};
            if (rowok && qfull[i]) {
                const float* g = f2b + (size_t)r * WW + goff[i];
                va = *(const v4f*)g;
                vb = *(const v4f*)(g + chs);
            } else if (rowok && qany[i]) {
                const float* g = f2b + (size_t)r * WW + goff[i];
                #pragma unroll
                for (int j = 0; j < 4; ++j) {
                    const int wj = w2v[i] + j;
                    if (wj >= 0 && wj < WW && (nqv[i] + j) < (WT + 2 * MAXD)) {
                        va[j] = g[j];
                        vb[j] = g[(size_t)chs + j];
                    }
                }
            }
            _Float16* dp = &Bf[dslot[i]];
            #pragma unroll
            for (int j = 0; j < 4; ++j)
                *(h2*)(dp + j * 16) = (h2){(_Float16)va[j], (_Float16)vb[j]};
        }
        __syncthreads();

        // ---- wave wv computes (h=h0+wv, dy=k-wv); wave-uniform guard ----
        const int dy = k - wv;
        if (dy >= 0 && dy < ND) {
            v8f acc0 = {};
            v8f acc1 = {};
            #pragma unroll
            for (int kc = 0; kc < 4; ++kc) {
                const _Float16* p0 = &Bf[(kc * 2 + hi) * BBLK + row16 * 16 + 8 * (row16 >> 3)];
                const _Float16* p1 = p0 + 272;   // cols 16..31: +16*16 + 8*2 halfs
                v16h b0, b1;
                #pragma unroll
                for (int t = 0; t < 16; ++t) { b0[t] = p0[t]; b1[t] = p1[t]; }
                acc0 = __builtin_amdgcn_wmma_f32_16x16x32_f16(
                           false, afrag[kc], false, b0, (short)0, acc0, false, false);
                acc1 = __builtin_amdgcn_wmma_f32_16x16x32_f16(
                           false, afrag[kc], false, b1, (short)0, acc1, false, false);
            }

            // spill C/D to padded G (conflict-free), same-wave in-order LDS
            float* G = &Glds[wv][0];
            #pragma unroll
            for (int r8 = 0; r8 < 8; ++r8) {
                G[(r8 + 8 * hi) * GS + row16]      = acc0[r8];
                G[(r8 + 8 * hi) * GS + 16 + row16] = acc1[r8];
            }

            // out[m][dx] = G[m][m+dx] * invc; hoisted offsets, coalesced stores
            float* op = outbase + (size_t)dy * (ND * HWSZ);
            #pragma unroll
            for (int i = 0; i < 4; ++i)
                op[oofs[i]] = G[gofs[i]] * invc;
            if (lane < 16)                       // tail: e = 128..143
                op[oofs[4]] = G[gofs[4]] * invc;
        }
        __syncthreads();   // protect Bf before next row's staging
    }
}

extern "C" void kernel_launch(void* const* d_in, const int* in_sizes, int n_in,
                              void* d_out, int out_size, void* d_ws, size_t ws_size,
                              hipStream_t stream) {
    const float* f1 = (const float*)d_in[0];
    const float* f2 = (const float*)d_in[1];
    float* out = (float*)d_out;
    (void)in_sizes; (void)n_in; (void)out_size; (void)d_ws; (void)ws_size;

    dim3 grid(WW / WT, HH / HB, NB);   // (12, 12, 8)
    dim3 block(256, 1, 1);             // 8 waves; wave w owns output row h0+w
    CorrelationLayer_48773648613915_kernel<<<grid, block, 0, stream>>>(f1, f2, out);
}
